// PyNeRFMLP_11587821765175
// MI455X (gfx1250) — matrix operand order repaired and checked by reading, not compiled, not run on hardware
//
#include <hip/hip_runtime.h>
#include <math.h>

// ---------------------------------------------------------------------------
// NeRF MLP forward on gfx1250 (CDNA5) using v_wmma_f32_16x16x32_f16.
// One wave (32 lanes) processes a tile of 16 points (WMMA M=16).
// Weights are packed to f16 in LDS in B-fragment order; activations bounce
// through per-wave LDS ping-pong buffers to perform the C->A transpose.
// Epilogues (ReLU + f32->f16) run packed: v_cvt_pk_f16_f32 + v_pk_max_num_f16.
// ---------------------------------------------------------------------------

#define HID 64
#define IN_DIM 32
#define WAVES_PER_BLOCK 4
#define BLOCK_THREADS (WAVES_PER_BLOCK * 32)

typedef __attribute__((ext_vector_type(16))) _Float16 v16h;
typedef __attribute__((ext_vector_type(2)))  _Float16 v2h;
typedef __attribute__((ext_vector_type(8)))  float    v8f;
typedef __attribute__((ext_vector_type(4)))  unsigned int u32x4;

union F16Frag { v16h h; u32x4 q[2]; };

__device__ __forceinline__ float relu1(float x) { return x > 0.f ? x : 0.f; }

__device__ __forceinline__ v8f wmma32(v16h a, v16h b, v8f c) {
  // D = A(16x32 f16) * B(32x16 f16) + C(16x16 f32)
  return __builtin_amdgcn_wmma_f32_16x16x32_f16(
      /*neg_a=*/false, a, /*neg_b=*/false, b,
      /*c_mod=*/(short)0, c, /*reuse_a=*/false, /*reuse_b=*/false);
}

// B fragment: weights stored column-major packed f16, W[ncols][KP].
// Lane layout (ISA 7.12.2): N = lane%16; elements 0..7 -> K = 8g..8g+7,
// elements 8..15 -> K = 16+8g..16+8g+7 (g = lane/16) (two ds_load_b128).
__device__ __forceinline__ v16h load_b(const _Float16* W, int KP, int t,
                                       int chunk, int lane) {
  int n = (t << 4) + (lane & 15);
  int g = lane >> 4;
  const u32x4* p = (const u32x4*)(W + n * KP + chunk * 32 + g * 8);
  F16Frag f;
  f.q[0] = p[0];
  f.q[1] = p[2];
  return f.h;
}

// Single-column B fragment (zero-padded 64->1 / 64->3 output layers).
__device__ __forceinline__ v16h load_b_col(const _Float16* Wcol, int chunk,
                                           int lane) {
  int g = lane >> 4;
  const u32x4* p = (const u32x4*)(Wcol + chunk * 32 + g * 8);
  F16Frag f;
  f.q[0] = p[0];
  f.q[1] = p[2];
  return f.h;
}

// A fragment from row-major f16 activations act[16][HID].
// Lane layout (ISA 7.12.2): M = lane%16; elements 0..7 -> K = 8g..8g+7,
// elements 8..15 -> K = 16+8g..16+8g+7 (g = lane/16) within a 32-K chunk.
__device__ __forceinline__ v16h load_a(const _Float16* act, int chunk,
                                       int lane) {
  int m = lane & 15, g = lane >> 4;
  const _Float16* base = act + m * HID + chunk * 32 + g * 8;
  F16Frag f;
  f.q[0] = *(const u32x4*)(base);       // K run 1 (16 bytes)
  f.q[1] = *(const u32x4*)(base + 16);  // K run 2 (+32 bytes)
  return f.h;
}

// Packed ReLU + f32->f16 + scatter a C fragment into the activation buffer.
// C layout: lane holds column N = 16t + lane%16, rows M = 8*(lane/16)+r.
// Pairs of rows share one dword: cvt_pk + pk_max, then lo/hi b16 stores.
__device__ __forceinline__ void store_act_relu(_Float16* act, int t, v8f c,
                                               int lane) {
  int n = (t << 4) + (lane & 15);
  int g = lane >> 4;
  const v2h z = {(_Float16)0.f, (_Float16)0.f};
#pragma unroll
  for (int r = 0; r < 8; r += 2) {
    v2h p;
    p[0] = (_Float16)c[r];
    p[1] = (_Float16)c[r + 1];
    p = __builtin_elementwise_max(p, z);  // v_pk_max_num_f16
    act[(8 * g + r) * HID + n]     = p[0];  // ds_store_b16
    act[(8 * g + r + 1) * HID + n] = p[1];  // ds_store_b16_d16_hi
  }
}

__device__ __forceinline__ v8f bias_frag(const float* b, int t, int lane) {
  float v = b[(t << 4) + (lane & 15)];
  v8f c;
#pragma unroll
  for (int r = 0; r < 8; ++r) c[r] = v;
  return c;
}

__device__ __forceinline__ void put4(v16h& h, int base, float4 f) {
  h[base + 0] = (_Float16)f.x;
  h[base + 1] = (_Float16)f.y;
  h[base + 2] = (_Float16)f.z;
  h[base + 3] = (_Float16)f.w;
}

__global__ __launch_bounds__(BLOCK_THREADS) void nerf_mlp_kernel(
    const float* __restrict__ pts, const float* __restrict__ vdirs,
    const float* __restrict__ dw0, const float* __restrict__ db0,
    const float* __restrict__ dw1, const float* __restrict__ db1,
    const float* __restrict__ dw2, const float* __restrict__ db2,
    const float* __restrict__ fw,  const float* __restrict__ fb,
    const float* __restrict__ cw0, const float* __restrict__ cb0,
    const float* __restrict__ cw1, const float* __restrict__ cb1,
    const float* __restrict__ cw2, const float* __restrict__ cb2,
    float* __restrict__ out, int npts) {
  // Weights, column-major packed f16 (B-fragment order).
  __shared__ __align__(16) _Float16 sWD0[64 * 32];   // 32 -> 64
  __shared__ __align__(16) _Float16 sWD1[64 * 64];   // 64 -> 64
  __shared__ __align__(16) _Float16 sWF [64 * 32];   // 32 -> 64
  __shared__ __align__(16) _Float16 sWC0[64 * 96];   // 67(pad 96) -> 64
  __shared__ __align__(16) _Float16 sWC1[64 * 64];   // 64 -> 64
  __shared__ __align__(16) _Float16 sWD2[64];        // 64 -> 1
  __shared__ __align__(16) _Float16 sWC2[3 * 64];    // 64 -> 3
  __shared__ float sB0[64], sB1[64], sFB[64], sCB0[64], sCB1[64];
  __shared__ float sCB2[4], sDB2[1];
  // Per-wave ping-pong activation tiles: [16 points][HID] f16.
  __shared__ __align__(16) _Float16 sAct[WAVES_PER_BLOCK][2][16 * HID];

  const int tid = threadIdx.x;

  // ---- one-time per block: fp32 weights -> packed f16 LDS ----
  {
    auto pack = [&](_Float16* dst, const float* W, int Kin, int KP, int ncols) {
      int total = ncols * KP;
      for (int i = tid; i < total; i += BLOCK_THREADS) {
        int n = i / KP, k = i - n * KP;
        dst[i] = (_Float16)((k < Kin) ? W[k * ncols + n] : 0.f);
      }
    };
    pack(sWD0, dw0, 32, 32, 64);
    pack(sWD1, dw1, 64, 64, 64);
    pack(sWF,  fw,  32, 32, 64);
    pack(sWC0, cw0, 67, 96, 64);  // rows 64..66 = viewdir weights, 67..95 = 0
    pack(sWC1, cw1, 64, 64, 64);
    pack(sWD2, dw2, 64, 64, 1);
    pack(sWC2, cw2, 64, 64, 3);
    for (int i = tid; i < 64; i += BLOCK_THREADS) {
      sB0[i] = db0[i]; sB1[i] = db1[i]; sFB[i] = fb[i];
      sCB0[i] = cb0[i]; sCB1[i] = cb1[i];
    }
    if (tid < 3) sCB2[tid] = cb2[tid];
    if (tid == 0) { sDB2[0] = db2[0]; sCB2[3] = 0.f; }
  }
  __syncthreads();

  const int lane = tid & 31;
  const int wave = tid >> 5;
  const int g = lane >> 4;      // half-wave id
  const int nrow = lane & 15;   // row (for A) / column (for B,C) id
  _Float16* actA = &sAct[wave][0][0];
  _Float16* actB = &sAct[wave][1][0];

  const int ntiles = npts >> 4;
  const int wstride = gridDim.x * WAVES_PER_BLOCK;
  for (int tile = blockIdx.x * WAVES_PER_BLOCK + wave; tile < ntiles;
       tile += wstride) {
    const int p0 = tile << 4;
    const int prow = p0 + nrow;

    // ---- input A fragment: encoded_pts[16 x 32] fp32 -> f16 ----
    F16Frag ain;
    {
      const float* rp = pts + (size_t)prow * IN_DIM + 8 * g;
      float4 f0 = *(const float4*)(rp);        // K = 8g..8g+3
      float4 f1 = *(const float4*)(rp + 4);    // K = 8g+4..8g+7
      float4 f2 = *(const float4*)(rp + 16);   // K = 16+8g..
      float4 f3 = *(const float4*)(rp + 20);
      put4(ain.h, 0, f0); put4(ain.h, 4, f1);
      put4(ain.h, 8, f2); put4(ain.h, 12, f3);
    }

    // ---- density L0: 32 -> 64, ReLU -> actA ----
#pragma unroll
    for (int t = 0; t < 4; ++t) {
      v8f c = bias_frag(sB0, t, lane);
      c = wmma32(ain.h, load_b(sWD0, 32, t, 0, lane), c);
      store_act_relu(actA, t, c, lane);
    }

    // ---- density L1: 64 -> 64, ReLU -> actB ----
    {
      v16h a0 = load_a(actA, 0, lane);
      v16h a1 = load_a(actA, 1, lane);
#pragma unroll
      for (int t = 0; t < 4; ++t) {
        v8f c = bias_frag(sB1, t, lane);
        c = wmma32(a0, load_b(sWD1, 64, t, 0, lane), c);
        c = wmma32(a1, load_b(sWD1, 64, t, 1, lane), c);
        store_act_relu(actB, t, c, lane);
      }
    }

    // ---- density L2: 64 -> 1 (cols 1..15 zero-padded), ReLU -> sigma ----
    {
      v16h a0 = load_a(actB, 0, lane);
      v16h a1 = load_a(actB, 1, lane);
      u32x4 zq = {0u, 0u, 0u, 0u};
      F16Frag zb; zb.q[0] = zq; zb.q[1] = zq;
      v16h b0 = zb.h, b1 = zb.h;
      float bv = (nrow == 0) ? sDB2[0] : 0.f;
      v8f c;
#pragma unroll
      for (int r = 0; r < 8; ++r) c[r] = bv;
      if (nrow == 0) {                 // only column 0 carries weights
        b0 = load_b_col(sWD2, 0, lane);
        b1 = load_b_col(sWD2, 1, lane);
      }
      c = wmma32(a0, b0, c);
      c = wmma32(a1, b1, c);
      if (nrow == 0) {
        float* sig = out + (size_t)npts * 3 + p0 + 8 * g;
#pragma unroll
        for (int r = 0; r < 8; ++r) sig[r] = relu1(c[r]);
      }
    }

    // ---- feature: 32 -> 64, ReLU -> actA (reuses ain) ----
#pragma unroll
    for (int t = 0; t < 4; ++t) {
      v8f c = bias_frag(sFB, t, lane);
      c = wmma32(ain.h, load_b(sWF, 32, t, 0, lane), c);
      store_act_relu(actA, t, c, lane);
    }

    // ---- normalized viewdirs as A fragment for color-L0 K-chunk 2 ----
    // Valid K = 64..66 -> elements 0..2 of lanes 0..15; everything else 0.
    F16Frag av;
    {
      u32x4 zq = {0u, 0u, 0u, 0u};
      av.q[0] = zq; av.q[1] = zq;
    }
    if (g == 0) {
      const float* vp = vdirs + (size_t)prow * 3;
      float vx = vp[0], vy = vp[1], vz = vp[2];
      float nrm = fmaxf(sqrtf(vx * vx + vy * vy + vz * vz), 1e-12f);
      float inv = 1.0f / nrm;
      av.h[0] = (_Float16)(vx * inv);
      av.h[1] = (_Float16)(vy * inv);
      av.h[2] = (_Float16)(vz * inv);
    }

    // ---- color L0: 67 (pad 96) -> 64, ReLU -> actB ----
    {
      v16h a0 = load_a(actA, 0, lane);
      v16h a1 = load_a(actA, 1, lane);
#pragma unroll
      for (int t = 0; t < 4; ++t) {
        v8f c = bias_frag(sCB0, t, lane);
        c = wmma32(a0,   load_b(sWC0, 96, t, 0, lane), c);
        c = wmma32(a1,   load_b(sWC0, 96, t, 1, lane), c);
        c = wmma32(av.h, load_b(sWC0, 96, t, 2, lane), c);
        store_act_relu(actB, t, c, lane);
      }
    }

    // ---- color L1: 64 -> 64, ReLU -> actA ----
    {
      v16h a0 = load_a(actB, 0, lane);
      v16h a1 = load_a(actB, 1, lane);
#pragma unroll
      for (int t = 0; t < 4; ++t) {
        v8f c = bias_frag(sCB1, t, lane);
        c = wmma32(a0, load_b(sWC1, 64, t, 0, lane), c);
        c = wmma32(a1, load_b(sWC1, 64, t, 1, lane), c);
        store_act_relu(actA, t, c, lane);
      }
    }

    // ---- color L2: 64 -> 3 (cols 3..15 zero-padded), sigmoid -> rgb ----
    {
      v16h a0 = load_a(actA, 0, lane);
      v16h a1 = load_a(actA, 1, lane);
      u32x4 zq = {0u, 0u, 0u, 0u};
      F16Frag zb; zb.q[0] = zq; zb.q[1] = zq;
      v16h b0 = zb.h, b1 = zb.h;
      float bv = (nrow < 3) ? sCB2[nrow] : 0.f;
      v8f c;
#pragma unroll
      for (int r = 0; r < 8; ++r) c[r] = bv;
      if (nrow < 3) {
        b0 = load_b_col(sWC2 + nrow * 64, 0, lane);
        b1 = load_b_col(sWC2 + nrow * 64, 1, lane);
      }
      c = wmma32(a0, b0, c);
      c = wmma32(a1, b1, c);
      if (nrow < 3) {
#pragma unroll
        for (int r = 0; r < 8; ++r) {
          float x = c[r];
          out[(size_t)(p0 + 8 * g + r) * 3 + nrow] = 1.f / (1.f + __expf(-x));
        }
      }
    }
  }
}

extern "C" void kernel_launch(void* const* d_in, const int* in_sizes, int n_in,
                              void* d_out, int out_size, void* d_ws,
                              size_t ws_size, hipStream_t stream) {
  const float* pts  = (const float*)d_in[0];
  const float* vdir = (const float*)d_in[1];
  const float* dw0  = (const float*)d_in[2];
  const float* db0  = (const float*)d_in[3];
  const float* dw1  = (const float*)d_in[4];
  const float* db1  = (const float*)d_in[5];
  const float* dw2  = (const float*)d_in[6];
  const float* db2  = (const float*)d_in[7];
  const float* fw   = (const float*)d_in[8];
  const float* fb   = (const float*)d_in[9];
  const float* cw0  = (const float*)d_in[10];
  const float* cb0  = (const float*)d_in[11];
  const float* cw1  = (const float*)d_in[12];
  const float* cb1  = (const float*)d_in[13];
  const float* cw2  = (const float*)d_in[14];
  const float* cb2  = (const float*)d_in[15];
  float* out = (float*)d_out;

  int npts = in_sizes[0] / IN_DIM;      // 2^21
  int ntiles = npts / 16;               // 131072 16-point tiles
  int blocks = (ntiles + WAVES_PER_BLOCK - 1) / WAVES_PER_BLOCK;
  if (blocks > 2048) blocks = 2048;     // grid-stride: ~16 tiles per wave

  nerf_mlp_kernel<<<blocks, BLOCK_THREADS, 0, stream>>>(
      pts, vdir, dw0, db0, dw1, db1, dw2, db2, fw, fb, cw0, cb0, cw1, cb1,
      cw2, cb2, out, npts);
}
